// Attention_59846074302777
// MI455X (gfx1250) — matrix operand order repaired and checked by reading, not compiled, not run on hardware
//
#include <hip/hip_runtime.h>
#include <hip/hip_bf16.h>

typedef __attribute__((ext_vector_type(16))) __bf16 v16bf;
typedef __attribute__((ext_vector_type(8)))  __bf16 v8bf;
typedef __attribute__((ext_vector_type(4)))  __bf16 v4bf;
typedef __attribute__((ext_vector_type(2)))  __bf16 v2bf;
typedef __attribute__((ext_vector_type(8)))  float  v8f;
typedef __attribute__((ext_vector_type(4)))  float  v4f;
typedef __attribute__((ext_vector_type(2)))  float  v2f;

#define D_MODEL       1024
#define T_SEQ         2048
#define B_BATCH       4
#define BLOCK_THREADS 512   // 16 waves (wave32)
#define QROWS_BLK     64    // query rows per workgroup (4 subtiles of 16)
#define KBLK          32    // keys per flash-attention step
#define DCHUNK        256   // d columns owned by each wave (1024/4)
#define VT_STRIDE     40    // padded key-stride of transposed V tile (16B-aligned rows)

// Combine two 16-byte LDS loads into one 16-element bf16 WMMA fragment.
__device__ __forceinline__ v16bf frag_cat(const __bf16* p0, const __bf16* p1) {
    v8bf lo = *(const v8bf*)p0;
    v8bf hi = *(const v8bf*)p1;
    return __builtin_shufflevector(lo, hi, 0, 1, 2, 3, 4, 5, 6, 7,
                                           8, 9, 10, 11, 12, 13, 14, 15);
}

__global__ __launch_bounds__(BLOCK_THREADS)
void attn_fa_bf16_wmma(const float* __restrict__ Q, const float* __restrict__ K,
                       const float* __restrict__ V, float* __restrict__ Out) {
    extern __shared__ char smem[];
    __bf16* Qs   = (__bf16*)smem;                          // 64 x 1024 bf16
    __bf16* Ks   = Qs + QROWS_BLK * D_MODEL;               // 32 x 1024 bf16 (row major)
    __bf16* Vt   = Ks + KBLK * D_MODEL;                    // 1024 x 40 bf16 (d-major)
    float*  Ss   = (float*)(Vt + D_MODEL * VT_STRIDE);     // 64 x 32 fp32 logits
    __bf16* Ps   = (__bf16*)(Ss + QROWS_BLK * KBLK);       // 64 x 32 bf16 probabilities
    float*  rowM = (float*)(Ps + QROWS_BLK * KBLK);        // 64 running row-max
    float*  rowL = rowM + QROWS_BLK;                       // 64 running row-sum
    float*  alphaS = rowL + QROWS_BLK;                     // 64 rescale factors

    const int tid   = threadIdx.x;
    const int wave  = tid >> 5;
    const int lane  = tid & 31;
    const int hi    = lane >> 4;         // half-wave select
    const int lan16 = lane & 15;
    const int qsub  = wave >> 2;         // 0..3 : which 16-row query subtile
    const int dbase = (wave & 3) * DCHUNK; // 0/256/512/768 : wave's d-chunk
    const int b     = blockIdx.y;
    const int r0    = blockIdx.x * QROWS_BLK;

    const float scale = 32.0f;           // 1/dk = sqrt(D_MODEL)

    // ---- stage Q block: fp32 global -> bf16 LDS (float4 vectorized) ----
    const float* Qg = Q + (size_t)(b * T_SEQ + r0) * D_MODEL;
    for (int idx = tid * 4; idx < QROWS_BLK * D_MODEL; idx += BLOCK_THREADS * 4) {
        v4f f = *(const v4f*)(Qg + idx);
        *(v4bf*)(Qs + idx) = __builtin_convertvector(f, v4bf);
    }

    if (tid < QROWS_BLK) { rowM[tid] = -__builtin_inff(); rowL[tid] = 0.0f; }

    const v8f vzero = {0.f, 0.f, 0.f, 0.f, 0.f, 0.f, 0.f, 0.f};
    v8f o[16];
#pragma unroll
    for (int nt = 0; nt < 16; ++nt) o[nt] = vzero;

    const int nblk = (r0 + QROWS_BLK - 1) / KBLK + 1;   // causal: keys <= r0+63

    for (int j = 0; j < nblk; ++j) {
        const int kb0 = j * KBLK;
        __syncthreads();   // previous iteration done reading Ks/Vt/Ps/Ss

        // ---- stage K block row-major (float4 vectorized) ----
        const float* Kg = K + (size_t)(b * T_SEQ + kb0) * D_MODEL;
        for (int idx = tid * 4; idx < KBLK * D_MODEL; idx += BLOCK_THREADS * 4) {
            v4f f = *(const v4f*)(Kg + idx);
            *(v4bf*)(Ks + idx) = __builtin_convertvector(f, v4bf);
        }
        // ---- stage V block transposed: (2-key x 2-d) micro-tiles ----
        const float* Vg = V + (size_t)(b * T_SEQ + kb0) * D_MODEL;
        for (int u = tid; u < (KBLK / 2) * (D_MODEL / 2); u += BLOCK_THREADS) {
            const int dp = u & (D_MODEL / 2 - 1);    // 0..511 (fast: coalesced along d)
            const int kp = u >> 9;                   // 0..15
            const int d0 = dp * 2, k0 = kp * 2;
            v2f va = *(const v2f*)(Vg + (size_t)k0 * D_MODEL + d0);
            v2f vb = *(const v2f*)(Vg + (size_t)(k0 + 1) * D_MODEL + d0);
            v2bf t0, t1;
            t0.x = (__bf16)va.x; t0.y = (__bf16)vb.x;   // row d0:   keys k0,k0+1
            t1.x = (__bf16)va.y; t1.y = (__bf16)vb.y;   // row d0+1: keys k0,k0+1
            *(v2bf*)(Vt + d0 * VT_STRIDE + k0)       = t0;
            *(v2bf*)(Vt + (d0 + 1) * VT_STRIDE + k0) = t1;
        }
        // zero the logit accumulator
        for (int idx = tid * 4; idx < QROWS_BLK * KBLK; idx += BLOCK_THREADS * 4)
            *(v4f*)(Ss + idx) = (v4f){0.f, 0.f, 0.f, 0.f};
        __syncthreads();

        // ---- gemm1: partial S(16x32) for this wave's subtile over its 256-d chunk ----
#pragma unroll
        for (int t = 0; t < 2; ++t) {
            v8f acc = vzero;
#pragma unroll
            for (int kk = 0; kk < 8; ++kk) {
                const int koff = dbase + kk * 32;
                // A fragment (16x32): lanes 0-15 K 0-7/16-23, lanes 16-31 K 8-15/24-31
                const __bf16* ap = Qs + (qsub * 16 + lan16) * D_MODEL + koff + hi * 8;
                v16bf a = frag_cat(ap, ap + 16);
                // B fragment (32x16): lane = key column; lanes 0-15 K 0-7/16-23, lanes 16-31 K 8-15/24-31
                const __bf16* bp = Ks + (t * 16 + lan16) * D_MODEL + koff + hi * 8;
                v16bf bm = frag_cat(bp, bp + 16);
                acc = __builtin_amdgcn_wmma_f32_16x16x32_bf16(
                          false, a, false, bm, (short)0, acc, false, false);
            }
            // C layout: VGPR jj -> row jj + 8*hi, lane&15 -> column
#pragma unroll
            for (int jj = 0; jj < 8; ++jj)
                atomicAdd(&Ss[(qsub * 16 + jj + hi * 8) * KBLK + t * 16 + lan16], acc[jj]);
        }
        __syncthreads();

        // ---- online softmax update: one thread per query row ----
        if (tid < QROWS_BLK) {
            const int grow = r0 + tid;
            float s[KBLK];
            float mt = -__builtin_inff();
#pragma unroll
            for (int c = 0; c < KBLK; ++c) {
                float v = Ss[tid * KBLK + c] * scale;
                if (kb0 + c > grow) v += -1000000000.0f;   // causal mask * NEG
                s[c] = v;
                mt = fmaxf(mt, v);
            }
            const float mold = rowM[tid];
            const float mnew = fmaxf(mold, mt);
            const float al   = __expf(mold - mnew);        // 0 when mold == -inf
            float sum = 0.0f;
#pragma unroll
            for (int c = 0; c < KBLK; ++c) {
                float p = __expf(s[c] - mnew);
                sum += p;
                Ps[tid * KBLK + c] = (__bf16)p;
            }
            rowL[tid]   = rowL[tid] * al + sum;
            rowM[tid]   = mnew;
            alphaS[tid] = al;
        }
        __syncthreads();

        // ---- rescale O, then gemm2: O += P(16x32) x V(32 x 256) ----
        float av[8];
#pragma unroll
        for (int jj = 0; jj < 8; ++jj) av[jj] = alphaS[qsub * 16 + jj + hi * 8];
#pragma unroll
        for (int nt = 0; nt < 16; ++nt) {
#pragma unroll
            for (int jj = 0; jj < 8; ++jj) o[nt][jj] *= av[jj];
        }
        const __bf16* pp = Ps + (qsub * 16 + lan16) * KBLK + hi * 8;
        v16bf a2 = frag_cat(pp, pp + 16);
#pragma unroll
        for (int nt = 0; nt < 16; ++nt) {
            // B fragment from transposed V: lane = d column, keys 0-7/16-23 (lanes 0-15), 8-15/24-31 (lanes 16-31)
            const __bf16* bp = Vt + (dbase + nt * 16 + lan16) * VT_STRIDE + hi * 8;
            v16bf b2 = frag_cat(bp, bp + 16);
            o[nt] = __builtin_amdgcn_wmma_f32_16x16x32_bf16(
                        false, a2, false, b2, (short)0, o[nt], false, false);
        }
    }

    // ---- normalize by row sums and store fp32 output ----
    float linv[8];
#pragma unroll
    for (int jj = 0; jj < 8; ++jj) linv[jj] = 1.0f / rowL[qsub * 16 + jj + hi * 8];
    float* Og = Out + (size_t)(b * T_SEQ + r0 + qsub * 16) * D_MODEL;
#pragma unroll
    for (int nt = 0; nt < 16; ++nt) {
#pragma unroll
        for (int jj = 0; jj < 8; ++jj)
            Og[(size_t)(jj + hi * 8) * D_MODEL + dbase + nt * 16 + lan16] =
                o[nt][jj] * linv[jj];
    }
}

extern "C" void kernel_launch(void* const* d_in, const int* in_sizes, int n_in,
                              void* d_out, int out_size, void* d_ws, size_t ws_size,
                              hipStream_t stream) {
    const float* Q = (const float*)d_in[0];
    const float* K = (const float*)d_in[1];
    const float* V = (const float*)d_in[2];
    // d_in[3] = dense causal mask; applied analytically inside the kernel.
    float* Out = (float*)d_out;

    const size_t smem =
        (size_t)(QROWS_BLK * D_MODEL + KBLK * D_MODEL + D_MODEL * VT_STRIDE) * sizeof(__bf16)
        + (size_t)QROWS_BLK * KBLK * sizeof(float)      // Ss
        + (size_t)QROWS_BLK * KBLK * sizeof(__bf16)     // Ps
        + (size_t)3 * QROWS_BLK * sizeof(float);        // rowM / rowL / alpha

    hipFuncSetAttribute(reinterpret_cast<const void*>(attn_fa_bf16_wmma),
                        hipFuncAttributeMaxDynamicSharedMemorySize, (int)smem);

    dim3 grid(T_SEQ / QROWS_BLK, B_BATCH);
    attn_fa_bf16_wmma<<<grid, BLOCK_THREADS, smem, stream>>>(Q, K, V, Out);
}
